// _Basexconv_82454782148820
// MI455X (gfx1250) — matrix operand order repaired and checked by reading, not compiled, not run on hardware
//
#include <hip/hip_runtime.h>

typedef __attribute__((ext_vector_type(16))) _Float16 v16h;
typedef __attribute__((ext_vector_type(8)))  _Float16 v8h;
typedef __attribute__((ext_vector_type(8)))  float    v8f;

// Problem constants (fixed by setup_inputs)
#define BB   16
#define NN   8192
#define PP   2048
#define KNB  16      // K neighbors kept
#define CD   64
#define CIN  64
#define CC   128     // CD + CIN
#define DMUL 4
#define COUT 256
#define SS   (BB*PP*KNB)   // 524288 samples (b,p,k)
#define NT   (BB*PP)       // 32768 points
#define QRSN (BB*3*PP)     // 98304 floats, first output

__device__ __forceinline__ float eluf(float x) {
    return x > 0.f ? x : (expf(x) - 1.f);
}

// ---- WMMA fragment loaders (wave32; contiguous k = 16h + s in both A and B, not the ISA table order) ----
// A: 16x32 f16 from row-major weights. Fragment order per lane is 16
// consecutive k values (kc+16h .. +15) -> two 128-bit loads.
__device__ __forceinline__ v16h load_a_frag(const _Float16* __restrict__ A,
                                            int row0, int lda, int kc, int lane) {
    int m = lane & 15, hh = lane >> 4;
    const _Float16* base = A + (size_t)(row0 + m) * lda + kc + 16 * hh;
    v8h lo = *(const v8h*)(base);
    v8h hi = *(const v8h*)(base + 8);
    return __builtin_shufflevector(lo, hi, 0, 1, 2, 3, 4, 5, 6, 7,
                                   8, 9, 10, 11, 12, 13, 14, 15);
}
// B: 32x16 f16 from chunk-packed storage: element (k,col) of a (Ktot x NTOT)
// matrix lives at ((k>>5)*NTOT + col)*32 + (k&31). A lane needs k = kc+16h..+15
// for its column -> one aligned 32-byte vector load.
__device__ __forceinline__ v16h load_b_frag_packed(const _Float16* __restrict__ Bp,
                                                   int kc, size_t ntot, int col0, int lane) {
    int n = lane & 15, hh = lane >> 4;
    return *(const v16h*)(Bp + ((size_t)(kc >> 5) * ntot + col0 + n) * 32 + 16 * hh);
}

// ---------------- Kernel 1: gather query points qrs -> d_out[0:QRSN] ----------------
__global__ __launch_bounds__(256) void k_qrs(const float* __restrict__ pts,
                                             const int* __restrict__ sind,
                                             float* __restrict__ qout) {
    int t = blockIdx.x * 256 + threadIdx.x;
    if (t >= NT) return;
    int b = t >> 11, p = t & 2047;
    int sid = sind[b * PP + p];
#pragma unroll
    for (int c = 0; c < 3; ++c)
        qout[((size_t)b * 3 + c) * PP + p] = pts[((size_t)b * 3 + c) * NN + sid];
}

// ---------------- Kernel 2: KNN (wave32 per query, top-32, take every 2nd) ----------------
__global__ __launch_bounds__(128) void k_knn(const float* __restrict__ pts,
                                             const int* __restrict__ sind,
                                             int* __restrict__ outIdx) {
    __shared__ float sd[4][32][32];
    __shared__ int   si[4][32][32];
    int lane = threadIdx.x & 31;
    int w    = threadIdx.x >> 5;
    int qid  = blockIdx.x * 4 + w;
    int b = qid >> 11, p = qid & 2047;
    const float* pb = pts + (size_t)b * 3 * NN;
    int sid = sind[b * PP + p];
    float qx = pb[sid], qy = pb[NN + sid], qz = pb[2 * NN + sid];
    float qq = qx * qx + qy * qy + qz * qz;
#pragma unroll
    for (int j = 0; j < 32; ++j) { sd[w][lane][j] = 3.0e38f; si[w][lane][j] = 0x7fffffff; }
    for (int n = lane; n < NN; n += 32) {
        float px = pb[n], py = pb[NN + n], pz = pb[2 * NN + n];
        float d = qq + (px * px + py * py + pz * pz) - 2.f * (qx * px + qy * py + qz * pz);
        float wrd = sd[w][lane][31]; int wri = si[w][lane][31];
        if (d < wrd || (d == wrd && n < wri)) {
            int j = 31;
            while (j > 0) {
                float pd = sd[w][lane][j - 1]; int pi = si[w][lane][j - 1];
                if (pd > d || (pd == d && pi > n)) { sd[w][lane][j] = pd; si[w][lane][j] = pi; --j; }
                else break;
            }
            sd[w][lane][j] = d; si[w][lane][j] = n;
        }
    }
    // merge 32 sorted lists: 32 rounds of cross-lane argmin
    int ptr = 0;
    for (int r = 0; r < 32; ++r) {
        float hd = (ptr < 32) ? sd[w][lane][ptr] : 3.4e38f;
        int   hi = (ptr < 32) ? si[w][lane][ptr] : 0x7fffffff;
        int   hl = lane;
#pragma unroll
        for (int off = 16; off >= 1; off >>= 1) {
            float od = __shfl_xor(hd, off, 32);
            int   oi = __shfl_xor(hi, off, 32);
            int   ol = __shfl_xor(hl, off, 32);
            if (od < hd || (od == hd && oi < hi)) { hd = od; hi = oi; hl = ol; }
        }
        if (lane == hl) ++ptr;
        if (((r & 1) == 0) && lane == 0) outIdx[(size_t)qid * KNB + (r >> 1)] = hi;
    }
}

// ---------------- Kernel 3: nn_pts_local (B,3,P,K) ----------------
__global__ __launch_bounds__(256) void k_local(const float* __restrict__ pts,
                                               const float* __restrict__ qrs,
                                               const int* __restrict__ idx,
                                               float* __restrict__ loc) {
    int t = blockIdx.x * 256 + threadIdx.x;
    if (t >= SS) return;
    int k = t & 15, bp = t >> 4, b = bp >> 11, p = bp & 2047;
    int i = idx[t];
#pragma unroll
    for (int c = 0; c < 3; ++c) {
        float v = pts[((size_t)b * 3 + c) * NN + i] - qrs[((size_t)b * 3 + c) * PP + p];
        loc[(((size_t)b * 3 + c) * PP + p) * KNB + k] = v;
    }
}

// ---------------- Kernel 4: h0 = BN(elu(w_d0 @ local)) -> packed f16 (64 x SS) ----------------
// thread -> (c = tid&63 fastest, s): packed stores are contiguous across lanes.
__global__ __launch_bounds__(256) void k_h0(const float* __restrict__ loc,
                                            const float* __restrict__ w_d0, const float* __restrict__ b_d0,
                                            const float* __restrict__ s_d0, const float* __restrict__ t_d0,
                                            _Float16* __restrict__ h0p) {
    size_t t = (size_t)blockIdx.x * 256 + threadIdx.x;   // over SS*64
    int c = (int)(t & 63);
    size_t s = t >> 6;
    int k = (int)(s & 15);
    int bp = (int)(s >> 4);
    int b = bp >> 11, p = bp & 2047;
    float l0 = loc[(((size_t)b * 3 + 0) * PP + p) * KNB + k];
    float l1 = loc[(((size_t)b * 3 + 1) * PP + p) * KNB + k];
    float l2 = loc[(((size_t)b * 3 + 2) * PP + p) * KNB + k];
    float a = b_d0[c] + w_d0[c * 3] * l0 + w_d0[c * 3 + 1] * l1 + w_d0[c * 3 + 2] * l2;
    a = eluf(a) * s_d0[c] + t_d0[c];
    h0p[((size_t)(c >> 5) * SS + s) * 32 + (c & 31)] = (_Float16)a;
}

// ---------------- weight f16 conversion ----------------
__global__ __launch_bounds__(256) void k_convert(const float* __restrict__ w_d1, _Float16* __restrict__ wd1h,
                                                 const float* __restrict__ w_pw, _Float16* __restrict__ wpwh) {
    int i = blockIdx.x * 256 + threadIdx.x;
    if (i < 64 * 64) wd1h[i] = (_Float16)w_d1[i];
    if (i < 256 * 512) wpwh[i] = (_Float16)w_pw[i];
}

// ---------------- Kernel 5: h1 = BN(elu(w_d1 @ h0)) via WMMA -> F[:, :64, :] ----------------
// 4 waves per block, one 16x16 M-tile per wave.
__global__ __launch_bounds__(128) void k_h1(const _Float16* __restrict__ wd1h,
                                            const _Float16* __restrict__ h0p,
                                            const float* __restrict__ b_d1,
                                            const float* __restrict__ s_d1, const float* __restrict__ t_d1,
                                            float* __restrict__ F) {
    int lane = threadIdx.x & 31;
    int mt   = threadIdx.x >> 5;          // 0..3 (CD/16)
    int s0   = blockIdx.x * 16;
    v8f acc = {};
#pragma unroll
    for (int kc = 0; kc < 64; kc += 32) {
        v16h a   = load_a_frag(wd1h, mt * 16, 64, kc, lane);
        v16h bfr = load_b_frag_packed(h0p, kc, (size_t)SS, s0, lane);
        acc = __builtin_amdgcn_wmma_f32_16x16x32_f16(false, a, false, bfr, (short)0, acc, false, false);
    }
    int n = lane & 15, col = s0 + n;
    int bp = col >> 4, k = col & 15;
#pragma unroll
    for (int r = 0; r < 8; ++r) {
        int m = mt * 16 + ((lane < 16) ? r : r + 8);
        float v = acc[r] + b_d1[m];
        v = eluf(v) * s_d1[m] + t_d1[m];
        F[((size_t)bp * CC + m) * KNB + k] = v;
    }
}

// ---------------- Kernel 6: gather fts -> F[:, 64:128, :] ----------------
__global__ __launch_bounds__(256) void k_fgather(const float* __restrict__ fts,
                                                 const int* __restrict__ idx,
                                                 float* __restrict__ F) {
    size_t t = (size_t)blockIdx.x * 256 + threadIdx.x;  // over NT*64*16
    if (t >= (size_t)NT * CIN * KNB) return;
    int k = (int)(t & 15);
    int c = (int)((t >> 4) & 63);
    int bp = (int)(t >> 10);
    int b = bp >> 11;
    int i = idx[(size_t)bp * KNB + k];
    F[((size_t)bp * CC + CIN + c) * KNB + k] = fts[((size_t)b * CIN + c) * NN + i];
}

// ---------------- Kernel 7: x0 = BN(elu(w_x0 . local)) -> (B,256,P) ----------------
__global__ __launch_bounds__(256) void k_x0(const float* __restrict__ loc,
                                            const float* __restrict__ w_x0, const float* __restrict__ b_x0,
                                            const float* __restrict__ s_x0, const float* __restrict__ t_x0,
                                            float* __restrict__ x0buf) {
    int p = blockIdx.x * 256 + threadIdx.x;
    int o = blockIdx.y, b = blockIdx.z;
    float acc = b_x0[o];
#pragma unroll
    for (int c = 0; c < 3; ++c)
#pragma unroll
        for (int k = 0; k < KNB; ++k)
            acc += w_x0[((size_t)o * 3 + c) * KNB + k] * loc[(((size_t)b * 3 + c) * PP + p) * KNB + k];
    acc = eluf(acc) * s_x0[o] + t_x0[o];
    x0buf[((size_t)b * 256 + o) * PP + p] = acc;
}

// ---------------- Kernels 8/9: scrambled-reshape group matmuls ----------------
// in[b,g,q,k] == buf[b, g*16 + (16q+k)>>11, (16q+k)&2047]  (row-major reshape of (B,256,P))
__global__ __launch_bounds__(256) void k_x1(const float* __restrict__ xin,
                                            const float* __restrict__ w, const float* __restrict__ bias,
                                            const float* __restrict__ sc, const float* __restrict__ sh,
                                            float* __restrict__ xout, int do_elu) {
    int q = blockIdx.x * 256 + threadIdx.x;
    int c = blockIdx.y, b = blockIdx.z;
    int g = c >> 4, j = c & 15;
    float acc = bias[c];
#pragma unroll
    for (int k = 0; k < KNB; ++k) {
        int f = (q << 4) + k;
        int c2 = (g << 4) + (f >> 11);
        int p2 = f & 2047;
        acc += w[(size_t)c * KNB + k] * xin[((size_t)b * 256 + c2) * PP + p2];
    }
    if (do_elu) acc = eluf(acc);
    acc = acc * sc[c] + sh[c];
    xout[((size_t)b * 256 + c) * PP + q] = acc;
}

// ---------------- Kernel 10: fts_X (per-point C x K * K x K) + depthwise -> packed dw f16 ----------------
__global__ __launch_bounds__(128) void k_ftsx_dw(const float* __restrict__ x2buf,
                                                 const float* __restrict__ F,
                                                 const float* __restrict__ w_dw, const float* __restrict__ b_dw,
                                                 const float* __restrict__ s_dw, const float* __restrict__ t_dw,
                                                 _Float16* __restrict__ dwp) {
    __shared__ float X[256];
    int bp = blockIdx.x;
    int b = bp >> 11, mp = bp & 2047;
    int tid = threadIdx.x;
#pragma unroll
    for (int e = tid; e < 256; e += 128) {
        int f = mp * 256 + e;                 // X[m,i,j] from flat reshape of (B,K,P,K)
        int c2 = f >> 11, p2 = f & 2047;
        X[e] = x2buf[((size_t)b * 256 + c2) * PP + p2];
    }
    __syncthreads();
    int c = tid;                               // channel 0..127
    float Fr[16], fX[16];
    const float* Fp = F + ((size_t)bp * CC + c) * KNB;
#pragma unroll
    for (int k = 0; k < 16; ++k) Fr[k] = Fp[k];
#pragma unroll
    for (int j = 0; j < 16; ++j) {
        float a = 0.f;
#pragma unroll
        for (int k = 0; k < 16; ++k) a += Fr[k] * X[k * 16 + j];
        fX[j] = a;
    }
#pragma unroll
    for (int d = 0; d < DMUL; ++d) {
        int cd = c * DMUL + d;
        float a = b_dw[cd];
#pragma unroll
        for (int k = 0; k < 16; ++k) a += w_dw[(size_t)cd * KNB + k] * fX[k];
        a = a * s_dw[cd] + t_dw[cd];
        dwp[((size_t)(cd >> 5) * NT + bp) * 32 + (cd & 31)] = (_Float16)a;
    }
}

// ---------------- Kernel 11: pointwise GEMM (WMMA) + bias + elu + BN -> d_out ----------------
// 4 waves per block, one 16x16 M-tile per wave.
__global__ __launch_bounds__(128) void k_pw(const _Float16* __restrict__ wpwh,
                                            const _Float16* __restrict__ dwp,
                                            const float* __restrict__ b_pw,
                                            const float* __restrict__ s_pw, const float* __restrict__ t_pw,
                                            float* __restrict__ outp) {
    int lane = threadIdx.x & 31;
    int mt   = blockIdx.y * 4 + (threadIdx.x >> 5);
    int nt   = blockIdx.x * 16;
    v8f acc = {};
#pragma unroll
    for (int kc = 0; kc < 512; kc += 32) {
        v16h a   = load_a_frag(wpwh, mt * 16, 512, kc, lane);
        v16h bfr = load_b_frag_packed(dwp, kc, (size_t)NT, nt, lane);
        acc = __builtin_amdgcn_wmma_f32_16x16x32_f16(false, a, false, bfr, (short)0, acc, false, false);
    }
    int n = lane & 15, col = nt + n;
    int b = col >> 11, p = col & 2047;
#pragma unroll
    for (int r = 0; r < 8; ++r) {
        int m = mt * 16 + ((lane < 16) ? r : r + 8);
        float v = acc[r] + b_pw[m];
        v = eluf(v) * s_pw[m] + t_pw[m];
        outp[((size_t)b * COUT + m) * PP + p] = v;
    }
}

extern "C" void kernel_launch(void* const* d_in, const int* in_sizes, int n_in,
                              void* d_out, int out_size, void* d_ws, size_t ws_size,
                              hipStream_t stream) {
    const float* pts  = (const float*)d_in[0];
    const float* fts  = (const float*)d_in[1];
    const int*   sind = (const int*)d_in[2];
    // d_in[3]=K, d_in[4]=D (constants, hardcoded)
    const float* w_d0 = (const float*)d_in[5];
    const float* b_d0 = (const float*)d_in[6];
    const float* s_d0 = (const float*)d_in[7];
    const float* t_d0 = (const float*)d_in[8];
    const float* w_d1 = (const float*)d_in[9];
    const float* b_d1 = (const float*)d_in[10];
    const float* s_d1 = (const float*)d_in[11];
    const float* t_d1 = (const float*)d_in[12];
    const float* w_x0 = (const float*)d_in[13];
    const float* b_x0 = (const float*)d_in[14];
    const float* s_x0 = (const float*)d_in[15];
    const float* t_x0 = (const float*)d_in[16];
    const float* w_x1 = (const float*)d_in[17];
    const float* b_x1 = (const float*)d_in[18];
    const float* s_x1 = (const float*)d_in[19];
    const float* t_x1 = (const float*)d_in[20];
    const float* w_x2 = (const float*)d_in[21];
    const float* b_x2 = (const float*)d_in[22];
    const float* s_x2 = (const float*)d_in[23];
    const float* t_x2 = (const float*)d_in[24];
    const float* w_dw = (const float*)d_in[25];
    const float* b_dw = (const float*)d_in[26];
    const float* s_dw = (const float*)d_in[27];
    const float* t_dw = (const float*)d_in[28];
    const float* w_pw = (const float*)d_in[29];
    const float* b_pw = (const float*)d_in[30];
    const float* s_pw = (const float*)d_in[31];
    const float* t_pw = (const float*)d_in[32];

    auto al = [](size_t x) { return (x + 255) & ~(size_t)255; };
    char* w = (char*)d_ws;
    size_t off = 0;
    int*      idxb  = (int*)(w + off);      off = al(off + (size_t)NT * KNB * 4);
    float*    loc   = (float*)(w + off);    off = al(off + (size_t)BB * 3 * PP * KNB * 4);
    _Float16* h0p   = (_Float16*)(w + off); off = al(off + (size_t)CD * SS * 2);
    _Float16* wd1h  = (_Float16*)(w + off); off = al(off + (size_t)64 * 64 * 2);
    _Float16* wpwh  = (_Float16*)(w + off); off = al(off + (size_t)256 * 512 * 2);
    float*    F     = (float*)(w + off);    off = al(off + (size_t)NT * CC * KNB * 4);
    float*    x0b   = (float*)(w + off);    off = al(off + (size_t)BB * 256 * PP * 4);
    float*    x1b   = (float*)(w + off);    off = al(off + (size_t)BB * 256 * PP * 4);
    float*    x2b   = (float*)(w + off);    off = al(off + (size_t)BB * 256 * PP * 4);
    _Float16* dwp   = (_Float16*)(w + off); off = al(off + (size_t)512 * NT * 2);
    if (ws_size < off) return;

    float* qrs_out  = (float*)d_out;
    float* conv_out = (float*)d_out + QRSN;

    k_qrs<<<NT / 256, 256, 0, stream>>>(pts, sind, qrs_out);
    k_knn<<<NT / 4, 128, 0, stream>>>(pts, sind, idxb);
    k_local<<<SS / 256, 256, 0, stream>>>(pts, qrs_out, idxb, loc);
    k_h0<<<(SS * 64) / 256, 256, 0, stream>>>(loc, w_d0, b_d0, s_d0, t_d0, h0p);
    k_convert<<<(256 * 512) / 256, 256, 0, stream>>>(w_d1, wd1h, w_pw, wpwh);
    k_h1<<<SS / 16, 128, 0, stream>>>(wd1h, h0p, b_d1, s_d1, t_d1, F);
    {
        size_t tot = (size_t)NT * CIN * KNB;
        k_fgather<<<(unsigned)((tot + 255) / 256), 256, 0, stream>>>(fts, idxb, F);
    }
    k_x0<<<dim3(PP / 256, 256, BB), 256, 0, stream>>>(loc, w_x0, b_x0, s_x0, t_x0, x0b);
    k_x1<<<dim3(PP / 256, 256, BB), 256, 0, stream>>>(x0b, w_x1, b_x1, s_x1, t_x1, x1b, 1);
    k_x1<<<dim3(PP / 256, 256, BB), 256, 0, stream>>>(x1b, w_x2, b_x2, s_x2, t_x2, x2b, 0);
    k_ftsx_dw<<<NT, 128, 0, stream>>>(x2b, F, w_dw, b_dw, s_dw, t_dw, dwp);
    k_pw<<<dim3(NT / 16, COUT / 16 / 4), 128, 0, stream>>>(wpwh, dwp, b_pw, s_pw, t_pw, conv_out);
}
